// Encoder_75900662055285
// MI455X (gfx1250) — matrix operand order repaired and checked by reading, not compiled, not run on hardware
//
#include <hip/hip_runtime.h>

// ---------------------------------------------------------------------------
// Fused 2-layer LSTM encoder for MI455X (gfx1250, wave32, WMMA).
//
//   out[b, 0:16] = relu( LSTM2( relu( LSTM1(x[b, 0:100, 0:18]) ) ).h_last )
//
// One wave owns 16 batch rows for the entire sequence:
//   * gates[16,64] computed as 4x v_wmma_f32_16x16x32_f16 tiles
//   * layer1: 2 WMMAs/gate (x-proj K=18 pad32, h-proj K=16 pad32)
//   * layer2: 1 WMMA/gate  (A = [relu(h1)|h2] packs K=32 exactly)
//   * c-state + nonlinearities stay in WMMA C/D register layout (v8f)
//   * h (C-layout) -> A-operand layout via 512B per-wave LDS transpose
// ---------------------------------------------------------------------------

typedef __attribute__((ext_vector_type(16))) _Float16 v16h;
typedef __attribute__((ext_vector_type(8)))  float    v8f;

#define T_SEQ  100
#define NFEAT  18
#define NHID   16
#define NBATCH 16384

__device__ __forceinline__ float sigm_f(float x) {
    return __builtin_amdgcn_rcpf(1.0f + __expf(-x));
}
__device__ __forceinline__ float tanh_f(float x) {
    float e = __expf(2.0f * x);
    return 1.0f - 2.0f * __builtin_amdgcn_rcpf(e + 1.0f);
}

__global__ __launch_bounds__(128, 1)
void lstm2_wmma_kernel(const float* __restrict__ x,
                       const float* __restrict__ Wih1, const float* __restrict__ Whh1,
                       const float* __restrict__ bih1, const float* __restrict__ bhh1,
                       const float* __restrict__ Wih2, const float* __restrict__ Whh2,
                       const float* __restrict__ bih2, const float* __restrict__ bhh2,
                       float* __restrict__ out)
{
    // 512 B per wave for the h C-layout -> A-layout transpose.
    __shared__ _Float16 hxch[4][16 * 16];

    const int lane  = threadIdx.x & 31;
    const int wv    = threadIdx.x >> 5;
    const int nloc  = lane & 15;     // column (gate unit / hidden unit) within tile
    const int lhalf = lane >> 4;     // lane half selects K/M sub-blocks
    const int m0    = (blockIdx.x * 4 + wv) * 16;   // first batch row of this tile

    _Float16* hb = hxch[wv];

    // ---- Preload weights into WMMA B-operand layout (held in VGPRs forever).
    // B (32x16 f16): VGPR j, lanes 0-15 hold K=(j/4)*16+(j%4)*2,+1; lanes 16-31 +8; N=lane%16.
    v16h b1x[4], b1h[4], b2[4];
    float bias1[4], bias2[4];
    #pragma unroll
    for (int g = 0; g < 4; ++g) {
        const int n = g * 16 + nloc;               // global gate index (i,f,g,o blocks of 16)
        #pragma unroll
        for (int j = 0; j < 8; ++j) {
            const int k0 = (j >> 2) * 16 + lhalf * 8 + (j & 3) * 2;
            const int k1 = k0 + 1;
            b1x[g][2*j]   = (k0 < NFEAT) ? (_Float16)Wih1[n * NFEAT + k0] : (_Float16)0.0f;
            b1x[g][2*j+1] = (k1 < NFEAT) ? (_Float16)Wih1[n * NFEAT + k1] : (_Float16)0.0f;
            b1h[g][2*j]   = (k0 < NHID)  ? (_Float16)Whh1[n * NHID  + k0] : (_Float16)0.0f;
            b1h[g][2*j+1] = (k1 < NHID)  ? (_Float16)Whh1[n * NHID  + k1] : (_Float16)0.0f;
            // layer 2: K 0..15 -> W_ih2 (input = relu(h1)), K 16..31 -> W_hh2 (recurrent h2)
            const float* w2a = (k0 < 16) ? (Wih2 + n * 16 + k0) : (Whh2 + n * 16 + (k0 - 16));
            const float* w2b = (k1 < 16) ? (Wih2 + n * 16 + k1) : (Whh2 + n * 16 + (k1 - 16));
            b2[g][2*j]   = (_Float16)(*w2a);
            b2[g][2*j+1] = (_Float16)(*w2b);
        }
        bias1[g] = bih1[n] + bhh1[n];   // reference folds b_ih + b_hh into xg
        bias2[g] = bih2[n] + bhh2[n];
    }

    // ---- LSTM state, all in registers.
    // C/D layout: VGPR r holds row m = r + 8*lhalf, column n = nloc.
    v8f c1 = {}, c2 = {};
    v8f h2v = {};
    // A-operand layout (16x32 f16): VGPR j: lanes0-15 K=(j/4)*16+(j%4)*2, lanes16-31 +8.
    v16h a1 = {};   // [h1 | 0]        for layer-1 h-projection
    v16h a2 = {};   // [relu(h1) | h2] for layer-2 combined projection
    #pragma unroll
    for (int i = 0; i < 16; ++i) { a1[i] = (_Float16)0.0f; a2[i] = (_Float16)0.0f; }

    const float* xrow = x + (size_t)(m0 + nloc) * (T_SEQ * NFEAT);

    for (int t = 0; t < T_SEQ; ++t) {
        const float* xt = xrow + t * NFEAT;

        // ---- A operand for x_t: 16x18 fp32 -> 16x32 f16 (zero padded K>=18).
        v16h ax;
        #pragma unroll
        for (int j = 0; j < 8; ++j) {
            const int kb = (j >> 2) * 16 + lhalf * 8 + (j & 3) * 2;
            ax[2*j]   = (kb     < NFEAT) ? (_Float16)xt[kb]     : (_Float16)0.0f;
            ax[2*j+1] = (kb + 1 < NFEAT) ? (_Float16)xt[kb + 1] : (_Float16)0.0f;
        }
        if (t + 1 < T_SEQ) __builtin_prefetch(xt + NFEAT, 0, 1);  // global_prefetch_b8

        // ---- Layer 1 gates: acc = bias + x*Wih1^T + h1*Whh1^T   (8 WMMAs)
        v8f acc[4];
        #pragma unroll
        for (int g = 0; g < 4; ++g) {
            v8f cb;
            #pragma unroll
            for (int r = 0; r < 8; ++r) cb[r] = bias1[g];
            cb = __builtin_amdgcn_wmma_f32_16x16x32_f16(false, ax, false, b1x[g],
                                                        (short)0, cb, false, false);
            acc[g] = __builtin_amdgcn_wmma_f32_16x16x32_f16(false, a1, false, b1h[g],
                                                            (short)0, cb, false, false);
        }

        // ---- Layer 1 elementwise (pure VALU in C/D layout).
        v8f h1;
        #pragma unroll
        for (int r = 0; r < 8; ++r) {
            const float iv = sigm_f(acc[0][r]);
            const float fv = sigm_f(acc[1][r]);
            const float gv = tanh_f(acc[2][r]);
            const float ov = sigm_f(acc[3][r]);
            const float cc = fv * c1[r] + iv * gv;
            c1[r] = cc;
            h1[r] = ov * tanh_f(cc);   // recurrence uses raw h1; relu only feeds layer 2
        }

        // ---- Transpose h1: C-layout -> A-layout via per-wave LDS (cross-lane).
        #pragma unroll
        for (int r = 0; r < 8; ++r)
            hb[(r + lhalf * 8) * 16 + nloc] = (_Float16)h1[r];
        asm volatile("s_wait_dscnt 0" ::: "memory");  // per-wave DS is in-order; fence compiler too
        #pragma unroll
        for (int j = 0; j < 4; ++j) {
            const int kb = lhalf * 8 + 2 * j;
            const _Float16 e0 = hb[nloc * 16 + kb];
            const _Float16 e1 = hb[nloc * 16 + kb + 1];
            a1[2*j]   = e0;
            a1[2*j+1] = e1;
            const _Float16 z = (_Float16)0.0f;
            a2[2*j]   = (e0 > z) ? e0 : z;   // relu(h1_seq) feeds layer 2
            a2[2*j+1] = (e1 > z) ? e1 : z;
        }
        asm volatile("" ::: "memory");  // keep loads above the h2 store below

        // ---- Layer 2 gates: acc = bias + [relu(h1)|h2] * [Wih2;Whh2]^T   (4 WMMAs)
        v8f acc2[4];
        #pragma unroll
        for (int g = 0; g < 4; ++g) {
            v8f cb;
            #pragma unroll
            for (int r = 0; r < 8; ++r) cb[r] = bias2[g];
            acc2[g] = __builtin_amdgcn_wmma_f32_16x16x32_f16(false, a2, false, b2[g],
                                                             (short)0, cb, false, false);
        }

        // ---- Layer 2 elementwise.
        #pragma unroll
        for (int r = 0; r < 8; ++r) {
            const float iv = sigm_f(acc2[0][r]);
            const float fv = sigm_f(acc2[1][r]);
            const float gv = tanh_f(acc2[2][r]);
            const float ov = sigm_f(acc2[3][r]);
            const float cc = fv * c2[r] + iv * gv;
            c2[r] = cc;
            h2v[r] = ov * tanh_f(cc);
        }

        // ---- Transpose h2 into the high-K half of a2 (recurrent input, no relu).
        #pragma unroll
        for (int r = 0; r < 8; ++r)
            hb[(r + lhalf * 8) * 16 + nloc] = (_Float16)h2v[r];
        asm volatile("s_wait_dscnt 0" ::: "memory");
        #pragma unroll
        for (int j = 0; j < 4; ++j) {
            const int kb = lhalf * 8 + 2 * j;     // K-16 within the high half
            a2[8 + 2*j]     = hb[nloc * 16 + kb];
            a2[8 + 2*j + 1] = hb[nloc * 16 + kb + 1];
        }
        asm volatile("" ::: "memory");  // keep loads above next iteration's h1 store
    }

    // ---- out = relu(h2_last), C/D layout -> [B,16] row major.
    #pragma unroll
    for (int r = 0; r < 8; ++r) {
        const int m = r + lhalf * 8;
        const float v = (h2v[r] > 0.0f) ? h2v[r] : 0.0f;
        out[(size_t)(m0 + m) * 16 + nloc] = v;
    }
}

extern "C" void kernel_launch(void* const* d_in, const int* in_sizes, int n_in,
                              void* d_out, int out_size, void* d_ws, size_t ws_size,
                              hipStream_t stream) {
    (void)in_sizes; (void)n_in; (void)out_size; (void)d_ws; (void)ws_size;
    const float* x    = (const float*)d_in[0];
    const float* Wih1 = (const float*)d_in[1];
    const float* Whh1 = (const float*)d_in[2];
    const float* bih1 = (const float*)d_in[3];
    const float* bhh1 = (const float*)d_in[4];
    const float* Wih2 = (const float*)d_in[5];
    const float* Whh2 = (const float*)d_in[6];
    const float* bih2 = (const float*)d_in[7];
    const float* bhh2 = (const float*)d_in[8];
    float* out = (float*)d_out;

    const int tiles  = NBATCH / 16;        // 1024 waves
    const int blocks = tiles / 4;          // 4 waves per 128-thread block
    lstm2_wmma_kernel<<<dim3(blocks), dim3(128), 0, stream>>>(
        x, Wih1, Whh1, bih1, bhh1, Wih2, Whh2, bih2, bhh2, out);
}
